// MambaLM_4475355922475
// MI455X (gfx1250) — matrix operand order repaired and checked by reading, not compiled, not run on hardware
//
#include <hip/hip_runtime.h>
#include <hip/hip_bf16.h>

// ---------------------------------------------------------------------------
// Mamba LM forward for gfx1250 (MI455X).
// Big GEMMs: v_wmma_f32_16x16x32_bf16 with B-tiles staged into LDS through the
// gfx1250 async copy path (global_load_async_to_lds_b128 + s_wait_asynccnt),
// double-buffered (static buffer indices, K-loop unrolled x2) so the next
// weight tile streams in while WMMA runs.
// Selective scan: register-resident per-channel recurrence, B/C via LDS.
// ---------------------------------------------------------------------------

#define L_SEQ   2048
#define D_MOD   768
#define ED      1536      // d_inner
#define NST     16        // d_state
#define DTR     48        // dt_rank
#define DBC_W   (DTR + 2*NST)   // 80
#define VOCAB_N 32000
#define NLAYERS 4

typedef __bf16 bf16;
typedef bf16  v16bf __attribute__((ext_vector_type(16)));
typedef bf16  v8bf  __attribute__((ext_vector_type(8)));
typedef float v8f   __attribute__((ext_vector_type(8)));

// ------------------------------ GEMM (WMMA) --------------------------------
// C[M,N] (f32) = A[M,K] (bf16, row-major) * W[N,K]^T (bf16, row-major)
//              (+ R[M,N] residual if R != nullptr)
// One wave -> 16x64 output strip (4 accumulators). Block = 8 waves = 128x64.
// B tile (64 rows x 32 k = 4KB) staged in LDS via async copy, double-buffered.
// Launch requirement here: M % 128 == 0, N % 64 == 0, K % 32 == 0.

__device__ __forceinline__ v16bf load_a_frag(const bf16* __restrict__ A,
                                             int lda, int m0, int k0, int lane) {
  const int hi = lane >> 4;          // 0: K{0..15}  1: K{16..31}
  const int m  = lane & 15;
  const bf16* p = A + (size_t)(m0 + m) * lda + k0 + hi * 16;
  v8bf lo = *(const v8bf*)(p);
  v8bf hv = *(const v8bf*)(p + 8);
  return __builtin_shufflevector(lo, hv, 0,1,2,3,4,5,6,7,8,9,10,11,12,13,14,15);
}

__global__ void __launch_bounds__(256)
gemm_bf16_wmma(const bf16* __restrict__ A, const bf16* __restrict__ W,
               const float* __restrict__ R, float* __restrict__ C,
               int M, int N, int K) {
  // LDS-staged B tile: [buf][n(0..63)][k(0..31)] bf16, row stride 32 halfs.
  __shared__ bf16 smemB[2][64 * 32];

  const int lane = threadIdx.x & 31;
  const int wave = threadIdx.x >> 5;
  const int m0 = blockIdx.y * 128 + wave * 16;
  const int n0 = blockIdx.x * 64;

  const int hi = lane >> 4;
  const int nn = lane & 15;

  // async B-tile copy: 256 threads x 16B = 4KB (row = tid>>2, 16B seg = tid&3)
  const int brow = threadIdx.x >> 2;
  const int bseg = threadIdx.x & 3;
  const bf16* gB0 = W + (size_t)(n0 + brow) * K + bseg * 8;
  const unsigned ldsa0 = (unsigned)(uintptr_t)(&smemB[0][brow * 32 + bseg * 8]);
  const unsigned ldsa1 = (unsigned)(uintptr_t)(&smemB[1][brow * 32 + bseg * 8]);

  auto issueB = [&](unsigned ldsa, int k0) {
    unsigned long long ga = (unsigned long long)(uintptr_t)(gB0 + k0);
    asm volatile("global_load_async_to_lds_b128 %0, %1, off"
                 :: "v"(ldsa), "v"(ga) : "memory");
  };

  v8f acc0, acc1, acc2, acc3;
  if (R != nullptr) {
#pragma unroll
    for (int r = 0; r < 8; ++r) {
      const size_t row = (size_t)(m0 + hi * 8 + r) * (size_t)N + n0 + nn;
      acc0[r] = R[row +  0];
      acc1[r] = R[row + 16];
      acc2[r] = R[row + 32];
      acc3[r] = R[row + 48];
    }
  } else {
#pragma unroll
    for (int r = 0; r < 8; ++r) { acc0[r] = 0.f; acc1[r] = 0.f; acc2[r] = 0.f; acc3[r] = 0.f; }
  }

  // 4 WMMAs against the LDS tile in buffer `bb`
  auto compute = [&](const bf16* bb, int k0) {
    v16bf a = load_a_frag(A, K, m0, k0, lane);
    v16bf b0 = *(const v16bf*)(bb +  0 * 32);
    acc0 = __builtin_amdgcn_wmma_f32_16x16x32_bf16(false, a, false, b0, (short)0, acc0, false, false);
    v16bf b1 = *(const v16bf*)(bb + 16 * 32);
    acc1 = __builtin_amdgcn_wmma_f32_16x16x32_bf16(false, a, false, b1, (short)0, acc1, false, false);
    v16bf b2 = *(const v16bf*)(bb + 32 * 32);
    acc2 = __builtin_amdgcn_wmma_f32_16x16x32_bf16(false, a, false, b2, (short)0, acc2, false, false);
    v16bf b3 = *(const v16bf*)(bb + 48 * 32);
    acc3 = __builtin_amdgcn_wmma_f32_16x16x32_bf16(false, a, false, b3, (short)0, acc3, false, false);
  };

  const bf16* bb0 = &smemB[0][(size_t)nn * 32 + hi * 16];
  const bf16* bb1 = &smemB[1][(size_t)nn * 32 + hi * 16];

  issueB(ldsa0, 0);                         // prologue: first B tile -> buf0

  int k0 = 0;
  for (; k0 + 64 <= K; k0 += 64) {          // two k-steps, static buffers
    asm volatile("s_wait_asynccnt 0x0" ::: "memory");
    __syncthreads();
    issueB(ldsa1, k0 + 32);                 // stream next tile -> buf1
    compute(bb0, k0);

    asm volatile("s_wait_asynccnt 0x0" ::: "memory");
    __syncthreads();
    if (k0 + 64 < K) issueB(ldsa0, k0 + 64);
    compute(bb1, k0 + 32);
  }
  if (k0 < K) {                             // K % 64 == 32 tail (unused here)
    asm volatile("s_wait_asynccnt 0x0" ::: "memory");
    __syncthreads();
    compute(bb0, k0);
  }

#pragma unroll
  for (int r = 0; r < 8; ++r) {
    const size_t row = (size_t)(m0 + hi * 8 + r) * (size_t)N + n0 + nn;
    C[row +  0] = acc0[r];
    C[row + 16] = acc1[r];
    C[row + 32] = acc2[r];
    C[row + 48] = acc3[r];
  }
}

// ----------------------------- small kernels -------------------------------

__global__ void cvt_bf16_kernel(const float* __restrict__ in, bf16* __restrict__ out, size_t n) {
  size_t i = (size_t)blockIdx.x * blockDim.x + threadIdx.x;
  if (i < n) out[i] = (bf16)in[i];
}

__global__ void embed_kernel(const int* __restrict__ tokens, const float* __restrict__ emb,
                             float* __restrict__ x) {
  const int t = blockIdx.x;
  const int tok = tokens[t];
  for (int i = threadIdx.x; i < D_MOD; i += blockDim.x)
    x[(size_t)t * D_MOD + i] = emb[(size_t)tok * D_MOD + i];
}

// rmsnorm over last dim -> bf16 output (for WMMA GEMM input)
__global__ void __launch_bounds__(256)
rmsnorm_bf16_kernel(const float* __restrict__ x, const float* __restrict__ w,
                    bf16* __restrict__ out, int D) {
  const int t = blockIdx.x;
  __shared__ float red[8];
  __shared__ float rtot;
  float s = 0.f;
  for (int i = threadIdx.x; i < D; i += blockDim.x) {
    float v = x[(size_t)t * D + i];
    s += v * v;
  }
#pragma unroll
  for (int o = 16; o > 0; o >>= 1) s += __shfl_xor(s, o, 32);
  if ((threadIdx.x & 31) == 0) red[threadIdx.x >> 5] = s;
  __syncthreads();
  if (threadIdx.x == 0) {
    float v = 0.f;
#pragma unroll
    for (int i = 0; i < 8; ++i) v += red[i];
    rtot = rsqrtf(v / (float)D + 1e-5f);
  }
  __syncthreads();
  const float r = rtot;
  for (int i = threadIdx.x; i < D; i += blockDim.x)
    out[(size_t)t * D + i] = (bf16)(x[(size_t)t * D + i] * r * w[i]);
}

// depthwise causal conv (k=4) + bias + silu on xb half of xz
__global__ void __launch_bounds__(256)
conv_silu_kernel(const float* __restrict__ xz, const float* __restrict__ cw,
                 const float* __restrict__ cb, float* __restrict__ xb) {
  const int e = blockIdx.x * blockDim.x + threadIdx.x;  // 0..ED-1
  const int t = blockIdx.y;
  float acc = cb[e];
#pragma unroll
  for (int j = 0; j < 4; ++j) {
    const int tt = t - 3 + j;
    if (tt >= 0) acc += cw[e * 4 + j] * xz[(size_t)tt * (2 * ED) + e];
  }
  const float s = acc / (1.f + __expf(-acc));           // silu
  xb[(size_t)t * ED + e] = s;
}

// dbc[t, 0..79] = xb[t,:] @ x_proj^T     (x_proj: 80 x 1536)
__global__ void __launch_bounds__(128)
xproj_kernel(const float* __restrict__ xb, const float* __restrict__ xp,
             float* __restrict__ dbc) {
  __shared__ float row[ED];
  const int t = blockIdx.x;
  for (int i = threadIdx.x; i < ED; i += blockDim.x)
    row[i] = xb[(size_t)t * ED + i];
  __syncthreads();
  for (int o = threadIdx.x; o < DBC_W; o += blockDim.x) {
    const float* wr = xp + (size_t)o * ED;
    float acc = 0.f;
    for (int k = 0; k < ED; ++k) acc += row[k] * wr[k];
    dbc[(size_t)t * DBC_W + o] = acc;
  }
}

// delta[t,e] = softplus(dbc[t,0..47] @ dt_w[e,:] + dt_b[e])
__global__ void __launch_bounds__(256)
delta_kernel(const float* __restrict__ dbc, const float* __restrict__ dtw,
             const float* __restrict__ dtb, float* __restrict__ delta) {
  __shared__ float d48[DTR];
  const int t = blockIdx.y;
  const int e = blockIdx.x * blockDim.x + threadIdx.x;
  if (threadIdx.x < DTR) d48[threadIdx.x] = dbc[(size_t)t * DBC_W + threadIdx.x];
  __syncthreads();
  float acc = dtb[e];
  const float* wr = dtw + (size_t)e * DTR;
#pragma unroll
  for (int k = 0; k < DTR; ++k) acc += d48[k] * wr[k];
  const float sp = (acc > 20.f) ? acc : __logf(1.f + __expf(acc));
  delta[(size_t)t * ED + e] = sp;
}

// selective scan: per-channel recurrence, h[16] in registers, B/C via LDS
__global__ void __launch_bounds__(256)
scan_kernel(const float* __restrict__ xb, const float* __restrict__ delta,
            const float* __restrict__ dbc, const float* __restrict__ A_log,
            const float* __restrict__ Dp, float* __restrict__ y) {
  const int e = blockIdx.x * blockDim.x + threadIdx.x;   // channel
  float Arow[NST];
#pragma unroll
  for (int n = 0; n < NST; ++n) Arow[n] = -__expf(A_log[(size_t)e * NST + n]);
  const float dp = Dp[e];
  float h[NST];
#pragma unroll
  for (int n = 0; n < NST; ++n) h[n] = 0.f;

  __shared__ float Bs[NST];
  __shared__ float Cs[NST];

  for (int t = 0; t < L_SEQ; ++t) {
    if (threadIdx.x < NST)
      Bs[threadIdx.x] = dbc[(size_t)t * DBC_W + DTR + threadIdx.x];
    else if (threadIdx.x < 2 * NST)
      Cs[threadIdx.x - NST] = dbc[(size_t)t * DBC_W + DTR + NST + (threadIdx.x - NST)];
    __syncthreads();

    const float dlt = delta[(size_t)t * ED + e];
    const float xv  = xb[(size_t)t * ED + e];
    const float dx  = dlt * xv;
    float acc = 0.f;
#pragma unroll
    for (int n = 0; n < NST; ++n) {
      const float dA = __expf(dlt * Arow[n]);
      h[n] = dA * h[n] + dx * Bs[n];
      acc += h[n] * Cs[n];
    }
    y[(size_t)t * ED + e] = acc + xv * dp;
    __syncthreads();
  }
}

// ym = bf16( y * silu(z) ),  z = xz[:, ED:2*ED]
__global__ void __launch_bounds__(256)
gate_kernel(const float* __restrict__ y, const float* __restrict__ xz,
            bf16* __restrict__ ym) {
  const int e = blockIdx.x * blockDim.x + threadIdx.x;
  const int t = blockIdx.y;
  const float z = xz[(size_t)t * (2 * ED) + ED + e];
  const float g = z / (1.f + __expf(-z));
  ym[(size_t)t * ED + e] = (bf16)(y[(size_t)t * ED + e] * g);
}

// ------------------------------- launcher ----------------------------------

extern "C" void kernel_launch(void* const* d_in, const int* in_sizes, int n_in,
                              void* d_out, int out_size, void* d_ws, size_t ws_size,
                              hipStream_t stream) {
  const int*   tokens    = (const int*)d_in[0];
  const float* embedding = (const float*)d_in[1];
  const float* norm_w    = (const float*)d_in[2];
  const float* in_proj   = (const float*)d_in[3];
  const float* conv_w    = (const float*)d_in[4];
  const float* conv_b    = (const float*)d_in[5];
  const float* x_proj    = (const float*)d_in[6];
  const float* dt_w      = (const float*)d_in[7];
  const float* dt_b      = (const float*)d_in[8];
  const float* A_log     = (const float*)d_in[9];
  const float* Dp        = (const float*)d_in[10];
  const float* out_proj  = (const float*)d_in[11];
  const float* norm_f_w  = (const float*)d_in[12];
  const float* lm_head   = (const float*)d_in[13];
  float* logits = (float*)d_out;

  // workspace carve-up (256B aligned)
  char* ws = (char*)d_ws;
  auto alloc = [&](size_t bytes) -> char* {
    char* p = ws;
    ws += (bytes + 255) & ~((size_t)255);
    return p;
  };
  float* x      = (float*)alloc((size_t)L_SEQ * D_MOD * 4);
  bf16*  xn_bf  = (bf16*) alloc((size_t)L_SEQ * D_MOD * 2);
  float* xz     = (float*)alloc((size_t)L_SEQ * 2 * ED * 4);
  float* xb     = (float*)alloc((size_t)L_SEQ * ED * 4);
  float* dbc    = (float*)alloc((size_t)L_SEQ * DBC_W * 4);
  float* dlt    = (float*)alloc((size_t)L_SEQ * ED * 4);
  float* yscan  = (float*)alloc((size_t)L_SEQ * ED * 4);
  bf16*  ym_bf  = (bf16*) alloc((size_t)L_SEQ * ED * 2);
  bf16*  w_in   = (bf16*) alloc((size_t)NLAYERS * 2 * ED * D_MOD * 2);
  bf16*  w_out  = (bf16*) alloc((size_t)NLAYERS * D_MOD * ED * 2);
  bf16*  w_lm   = (bf16*) alloc((size_t)VOCAB_N * D_MOD * 2);

  auto cvt = [&](const float* src, bf16* dst, size_t n) {
    cvt_bf16_kernel<<<(unsigned)((n + 255) / 256), 256, 0, stream>>>(src, dst, n);
  };
  auto gemm = [&](const bf16* A, const bf16* W, const float* R, float* C,
                  int M, int N, int K) {
    dim3 grid((unsigned)(N / 64), (unsigned)(M / 128));
    gemm_bf16_wmma<<<grid, 256, 0, stream>>>(A, W, R, C, M, N, K);
  };

  // weight precision conversion (deterministic; done every call)
  cvt(in_proj,  w_in,  (size_t)NLAYERS * 2 * ED * D_MOD);
  cvt(out_proj, w_out, (size_t)NLAYERS * D_MOD * ED);
  cvt(lm_head,  w_lm,  (size_t)VOCAB_N * D_MOD);

  // embedding gather
  embed_kernel<<<L_SEQ, 256, 0, stream>>>(tokens, embedding, x);

  const dim3 gridE(ED / 256, L_SEQ);   // (6, 2048)

  for (int l = 0; l < NLAYERS; ++l) {
    const bf16*  Wi  = w_in  + (size_t)l * 2 * ED * D_MOD;
    const bf16*  Wo  = w_out + (size_t)l * D_MOD * ED;
    const float* cwl = conv_w + (size_t)l * ED * 4;
    const float* cbl = conv_b + (size_t)l * ED;
    const float* xpl = x_proj + (size_t)l * DBC_W * ED;
    const float* dwl = dt_w   + (size_t)l * ED * DTR;
    const float* dbl = dt_b   + (size_t)l * ED;
    const float* Al  = A_log  + (size_t)l * ED * NST;
    const float* Dl  = Dp     + (size_t)l * ED;

    // xn = rmsnorm(x) -> bf16
    rmsnorm_bf16_kernel<<<L_SEQ, 256, 0, stream>>>(x, norm_w + (size_t)l * D_MOD,
                                                   xn_bf, D_MOD);
    // xz = xn @ in_proj^T   (2048 x 3072)
    gemm(xn_bf, Wi, nullptr, xz, L_SEQ, 2 * ED, D_MOD);
    // xb = silu(causal depthwise conv(xz[:, :ED]) + b)
    conv_silu_kernel<<<gridE, 256, 0, stream>>>(xz, cwl, cbl, xb);
    // dbc = xb @ x_proj^T   (2048 x 80)
    xproj_kernel<<<L_SEQ, 128, 0, stream>>>(xb, xpl, dbc);
    // delta = softplus(dbc[:, :48] @ dt_w^T + dt_b)
    delta_kernel<<<gridE, 256, 0, stream>>>(dbc, dwl, dbl, dlt);
    // selective scan (sequential in L, parallel over channels)
    scan_kernel<<<ED / 256, 256, 0, stream>>>(xb, dlt, dbc, Al, Dl, yscan);
    // ym = bf16(y * silu(z))
    gate_kernel<<<gridE, 256, 0, stream>>>(yscan, xz, ym_bf);
    // x = x + ym @ out_proj^T   (residual fused into WMMA accumulator init)
    gemm(ym_bf, Wo, x, x, L_SEQ, D_MOD, ED);
  }

  // final norm + lm_head
  rmsnorm_bf16_kernel<<<L_SEQ, 256, 0, stream>>>(x, norm_f_w, xn_bf, D_MOD);
  gemm(xn_bf, w_lm, nullptr, logits, L_SEQ, VOCAB_N, D_MOD);

  (void)in_sizes; (void)n_in; (void)out_size; (void)ws_size;
}
